// MotionLSTM_56521769615776
// MI455X (gfx1250) — matrix operand order repaired and checked by reading, not compiled, not run on hardware
//
#include <hip/hip_runtime.h>
#include <hip/hip_bf16.h>
#include <math.h>

typedef _Float16 half_t;
typedef __attribute__((ext_vector_type(16))) _Float16 v16h;
typedef __attribute__((ext_vector_type(8)))  _Float16 v8h;
typedef __attribute__((ext_vector_type(8)))  float    v8f;

#define B_      2
#define N_      4096
#define C_      256
#define KNN_K   16
#define KSTEPS_BIG 17      // padded K = 544 for I/F/O/C1 (X rows: 4 grp, 4 zero, 256 nn, 256 feat, 24 zero)
#define KSTEPS_C0  9       // padded K = 288 for C0 path
#define CSTR    552        // LDS column stride in halfs (>=544, 16-lane bank-conflict free)
#define ANCH    2          // anchors per GEMM block (32 columns)
#define COLS    (ANCH*16)
#define TQ      256        // query tile for KNN

// ---------------- [B,C,N] f32 -> [B,N,Ctot] f16 (with column offset for concat) ----------
__global__ void transpose_cast_kernel(const float* __restrict__ src, half_t* __restrict__ dst,
                                      int C, int dstStride, int dstOff)
{
    int i = blockIdx.x * blockDim.x + threadIdx.x;      // over B_*C*N_
    int n = i % N_;
    int c = (i / N_) % C;
    int b = i / (N_ * C);
    dst[((size_t)(b * N_ + n)) * dstStride + dstOff + c] = (half_t)src[i];
}

// ---------------- pack W [256 x Kw] f32 into WMMA 16-bit A-fragment order ----------------
// Linear layout: ((rowtile*ksteps + ks)*32 + lane)*16 + elem  -> two b128 loads per fragment.
__global__ void pack_w_kernel(const float* __restrict__ W, half_t* __restrict__ dst,
                              int ksteps, int Kw)
{
    int i = blockIdx.x * blockDim.x + threadIdx.x;
    int e    = i & 15;
    int lane = (i >> 4) & 31;
    int rk   = i >> 9;
    int ks   = rk % ksteps;
    int rt   = rk / ksteps;
    if (rt >= 16) return;
    int M = rt * 16 + (lane & 15);
    int v = e >> 1, h = e & 1;
    // ISA 16-bit A 16x32 layout: lanes<16 K={0..7,16..23}, lanes>=16 K={8..15,24..31}
    int kk = (v < 4 ? 2 * v : 16 + 2 * (v - 4)) + h + 8 * (lane >> 4);
    int r  = ks * 32 + kk;                       // padded X row
    int c  = (r < 4) ? r : ((r < 8) ? -1 : (r - 4));  // X row -> W column
    if (c >= Kw) c = -1;
    dst[i] = (half_t)((c >= 0) ? W[M * Kw + c] : 0.0f);
}

// ---------------- KNN: top-16 neighbors per anchor + relative-group features -------------
__global__ __launch_bounds__(128)
void knn_kernel(const float* __restrict__ p0, const float* __restrict__ p1,
                int* __restrict__ knn, half_t* __restrict__ group4)
{
    __shared__ float Qx[TQ], Qy[TQ], Qz[TQ];
    const int gid = blockIdx.x * 128 + threadIdx.x;     // N_ % 128 == 0 -> same b per block
    const int b = gid / N_;
    const int m = gid % N_;
    const float ax = p1[((size_t)b * 3 + 0) * N_ + m];
    const float ay = p1[((size_t)b * 3 + 1) * N_ + m];
    const float az = p1[((size_t)b * 3 + 2) * N_ + m];
    float bd[KNN_K]; int bi[KNN_K];
#pragma unroll
    for (int s = 0; s < KNN_K; ++s) { bd[s] = 3.4e38f; bi[s] = 0; }
    for (int t0 = 0; t0 < N_; t0 += TQ) {
        __syncthreads();
        for (int i = threadIdx.x; i < TQ; i += 128) {
            Qx[i] = p0[((size_t)b * 3 + 0) * N_ + t0 + i];
            Qy[i] = p0[((size_t)b * 3 + 1) * N_ + t0 + i];
            Qz[i] = p0[((size_t)b * 3 + 2) * N_ + t0 + i];
        }
        __syncthreads();
        for (int i = 0; i < TQ; ++i) {
            const float dx = Qx[i] - ax, dy = Qy[i] - ay, dz = Qz[i] - az;
            const float d = dx * dx + dy * dy + dz * dz;
            if (d < bd[KNN_K - 1]) {                // branchless sorted insert (static idx)
                float nd = d; int ni = t0 + i;
#pragma unroll
                for (int s = 0; s < KNN_K; ++s) {
                    const bool  sw = nd < bd[s];
                    const float td = sw ? nd : bd[s];
                    const int   ti = sw ? ni : bi[s];
                    const float od = bd[s]; const int oi = bi[s];
                    nd = sw ? od : nd;  ni = sw ? oi : ni;
                    bd[s] = td;         bi[s] = ti;
                }
            }
        }
    }
#pragma unroll
    for (int s = 0; s < KNN_K; ++s) {
        const int j = bi[s];
        knn[((size_t)(b * N_ + m)) * KNN_K + s] = j;
        const float dx = p0[((size_t)b * 3 + 0) * N_ + j] - ax;
        const float dy = p0[((size_t)b * 3 + 1) * N_ + j] - ay;
        const float dz = p0[((size_t)b * 3 + 2) * N_ + j] - az;
        half_t* gp = &group4[(((size_t)(b * N_ + m)) * KNN_K + s) * 4];
        gp[0] = (half_t)dx; gp[1] = (half_t)dy; gp[2] = (half_t)dz;
        gp[3] = (half_t)sqrtf(bd[s]);
    }
}

// ---------------- fused gather + WMMA GEMM + max-over-k per gate -------------------------
__global__ __launch_bounds__(256)
void gate_gemm_kernel(const half_t* __restrict__ H0h, const half_t* __restrict__ C0h,
                      const half_t* __restrict__ feath,
                      const half_t* __restrict__ group4, const int* __restrict__ knn,
                      const half_t* __restrict__ Wpack, const half_t* __restrict__ WpackC0,
                      float* __restrict__ G)
{
    __shared__ half_t Bs[COLS * CSTR];
    const int t    = threadIdx.x;
    const int m0   = blockIdx.x * ANCH;
    const int gate = blockIdx.y;          // 0=I 1=F 2=O 3=C1 4=C0
    const int b    = blockIdx.z;
    const bool big = (gate < 4);
    const int ksteps = big ? KSTEPS_BIG : KSTEPS_C0;
    const half_t* __restrict__ src = big ? H0h : C0h;

    // ---- stage B panel: 32 columns (2 anchors x 16 neighbors), column-major in LDS ----
    {
        const int col = t >> 3;           // 8 threads per column
        const int q   = t & 7;
        const int anchor = m0 + (col >> 4);
        const int nb     = col & 15;
        const int j = knn[((size_t)(b * N_ + anchor)) * KNN_K + nb];
        half_t* dst = &Bs[col * CSTR];
        if (q == 0) {                     // rows 0..3 group, rows 4..7 zero
            *(uint2*)dst = *(const uint2*)&group4[(((size_t)(b * N_ + anchor)) * KNN_K + nb) * 4];
            uint2 z2 = {0u, 0u};
            *(uint2*)(dst + 4) = z2;
        }
        {                                 // rows 8..263: gathered H0/C0 column (contiguous)
            const float4* s4 = (const float4*)(src + ((size_t)(b * N_ + j)) * C_ + q * 32);
            float4* d4 = (float4*)(dst + 8 + q * 32);
#pragma unroll
            for (int i = 0; i < 4; ++i) d4[i] = s4[i];
        }
        float4 z4 = {0.f, 0.f, 0.f, 0.f};
        if (big) {                        // rows 264..519: broadcast per-anchor features
            const float4* s4 = (const float4*)(feath + ((size_t)(b * N_ + anchor)) * C_ + q * 32);
            float4* d4 = (float4*)(dst + 264 + q * 32);
#pragma unroll
            for (int i = 0; i < 4; ++i) d4[i] = s4[i];
            if (q == 1) { float4* z = (float4*)(dst + 520); z[0] = z4; z[1] = z4; z[2] = z4; }
        } else {
            if (q == 1) { float4* z = (float4*)(dst + 264); z[0] = z4; z[1] = z4; z[2] = z4; }
        }
    }
    __syncthreads();

    const int wave   = t >> 5;
    const int lane   = t & 31;
    const int half16 = lane >> 4;
    const int ln     = lane & 15;
    const half_t* __restrict__ wp =
        big ? (Wpack + (size_t)gate * (16 * KSTEPS_BIG * 512)) : WpackC0;

    v8f acc[2][ANCH] = {};
    for (int ks = 0; ks < ksteps; ++ks) {
        // A fragments pre-packed in hardware layout: 2x global_load_b128 each
        v16h a0 = *(const v16h*)(wp + (((size_t)(2 * wave)     * ksteps + ks) * 32 + lane) * 16);
        v16h a1 = *(const v16h*)(wp + (((size_t)(2 * wave + 1) * ksteps + ks) * 32 + lane) * 16);
#pragma unroll
        for (int ct = 0; ct < ANCH; ++ct) {
            // B fragment: lane ln holds col N=ln, K = 8*half16 + j (j<8), 16 + 8*half16 + (j-8) (j>=8)
            const half_t* bp = &Bs[(ct * 16 + ln) * CSTR + ks * 32 + 8 * half16];
            v8h lo = *(const v8h*)bp;
            v8h hi = *(const v8h*)(bp + 16);
            v16h bb = __builtin_shufflevector(lo, hi, 0,1,2,3,4,5,6,7,8,9,10,11,12,13,14,15);
            acc[0][ct] = __builtin_amdgcn_wmma_f32_16x16x32_f16(
                false, a0, false, bb, (short)0, acc[0][ct], false, false);
            acc[1][ct] = __builtin_amdgcn_wmma_f32_16x16x32_f16(
                false, a1, false, bb, (short)0, acc[1][ct], false, false);
        }
    }

    // ---- max over the 16 neighbor columns (one anchor per 16x16 tile) + store ----
#pragma unroll
    for (int i = 0; i < 2; ++i) {
        const int otile = wave * 32 + i * 16;
#pragma unroll
        for (int ct = 0; ct < ANCH; ++ct) {
            const int m = m0 + ct;
#pragma unroll
            for (int r = 0; r < 8; ++r) {
                float v = acc[i][ct][r];
                v = fmaxf(v, __shfl_xor(v, 1, 16));
                v = fmaxf(v, __shfl_xor(v, 2, 16));
                v = fmaxf(v, __shfl_xor(v, 4, 16));
                v = fmaxf(v, __shfl_xor(v, 8, 16));
                if (ln == 0) {
                    const int o = otile + r + 8 * half16;   // C/D layout: M = r + 8*(lane>=16)
                    G[(((size_t)gate * B_ + b) * C_ + o) * N_ + m] = v;
                }
            }
        }
    }
}

// ---------------- LSTM combine ----------------------------------------------------------
__global__ void combine_kernel(const float* __restrict__ G,
                               const float* __restrict__ bI, const float* __restrict__ bF,
                               const float* __restrict__ bO, const float* __restrict__ bC0,
                               const float* __restrict__ bC1, float* __restrict__ out)
{
    size_t i = (size_t)blockIdx.x * blockDim.x + threadIdx.x;   // over B_*C_*N_
    int o = (int)((i / N_) % C_);
    size_t gs = (size_t)B_ * C_ * N_;
    float vI  = G[0 * gs + i] + bI[o];
    float vF  = G[1 * gs + i] + bF[o];
    float vO  = G[2 * gs + i] + bO[o];
    float vC1 = G[3 * gs + i] + bC1[o];
    float vC0 = G[4 * gs + i] + bC0[o];
    float gI = 1.f / (1.f + expf(-vI));
    float gF = 1.f / (1.f + expf(-vF));
    float gO = 1.f / (1.f + expf(-vO));
    float c1 = gF * vC0 + gI * tanhf(vC1);
    out[i]      = gO * tanhf(c1);   // H1
    out[gs + i] = c1;               // C1
}

// ---------------- host launch ------------------------------------------------------------
extern "C" void kernel_launch(void* const* d_in, const int* in_sizes, int n_in,
                              void* d_out, int out_size, void* d_ws, size_t ws_size,
                              hipStream_t stream)
{
    const float* H0        = (const float*)d_in[0];
    const float* C0        = (const float*)d_in[1];
    const float* points0   = (const float*)d_in[2];
    const float* points1   = (const float*)d_in[3];
    const float* contents1 = (const float*)d_in[4];
    const float* motions1  = (const float*)d_in[5];
    const float* W_I  = (const float*)d_in[6];
    const float* b_I  = (const float*)d_in[7];
    const float* W_F  = (const float*)d_in[8];
    const float* b_F  = (const float*)d_in[9];
    const float* W_O  = (const float*)d_in[10];
    const float* b_O  = (const float*)d_in[11];
    const float* W_C0 = (const float*)d_in[12];
    const float* b_C0 = (const float*)d_in[13];
    const float* W_C1 = (const float*)d_in[14];
    const float* b_C1 = (const float*)d_in[15];

    char* ws = (char*)d_ws;
    size_t off = 0;
    auto alloc = [&](size_t bytes) {
        char* p = ws + off;
        off += (bytes + 255) & ~(size_t)255;
        return p;
    };
    half_t* H0h     = (half_t*)alloc((size_t)B_ * N_ * C_ * 2);
    half_t* C0h     = (half_t*)alloc((size_t)B_ * N_ * C_ * 2);
    half_t* feath   = (half_t*)alloc((size_t)B_ * N_ * C_ * 2);
    half_t* group4  = (half_t*)alloc((size_t)B_ * N_ * KNN_K * 4 * 2);
    int*    knnIdx  = (int*)   alloc((size_t)B_ * N_ * KNN_K * 4);
    half_t* Wpack   = (half_t*)alloc((size_t)4 * 16 * KSTEPS_BIG * 512 * 2);
    half_t* WpackC0 = (half_t*)alloc((size_t)16 * KSTEPS_C0 * 512 * 2);
    float*  G       = (float*) alloc((size_t)5 * B_ * C_ * N_ * 4);

    // f32 [B,C,N] -> f16 [B,N,C] (feat = concat(contents, motions))
    transpose_cast_kernel<<<(B_ * C_ * N_) / 256, 256, 0, stream>>>(H0, H0h, C_, C_, 0);
    transpose_cast_kernel<<<(B_ * C_ * N_) / 256, 256, 0, stream>>>(C0, C0h, C_, C_, 0);
    transpose_cast_kernel<<<(B_ * 128 * N_) / 256, 256, 0, stream>>>(contents1, feath, 128, C_, 0);
    transpose_cast_kernel<<<(B_ * 128 * N_) / 256, 256, 0, stream>>>(motions1, feath, 128, C_, 128);

    // weight repack into WMMA A-fragment layout
    const int bigTot = 16 * KSTEPS_BIG * 512;
    pack_w_kernel<<<bigTot / 256, 256, 0, stream>>>(W_I,  Wpack + 0 * (size_t)bigTot, KSTEPS_BIG, 516);
    pack_w_kernel<<<bigTot / 256, 256, 0, stream>>>(W_F,  Wpack + 1 * (size_t)bigTot, KSTEPS_BIG, 516);
    pack_w_kernel<<<bigTot / 256, 256, 0, stream>>>(W_O,  Wpack + 2 * (size_t)bigTot, KSTEPS_BIG, 516);
    pack_w_kernel<<<bigTot / 256, 256, 0, stream>>>(W_C1, Wpack + 3 * (size_t)bigTot, KSTEPS_BIG, 516);
    const int c0Tot = 16 * KSTEPS_C0 * 512;
    pack_w_kernel<<<c0Tot / 256, 256, 0, stream>>>(W_C0, WpackC0, KSTEPS_C0, 260);

    // KNN
    knn_kernel<<<(B_ * N_) / 128, 128, 0, stream>>>(points0, points1, knnIdx, group4);

    // fused gather + GEMM + max over k, one gate per grid.y
    dim3 grid(N_ / ANCH, 5, B_);
    gate_gemm_kernel<<<grid, 256, 0, stream>>>(H0h, C0h, feath, group4, knnIdx, Wpack, WpackC0, G);

    // LSTM combine -> (H1, C1)
    combine_kernel<<<(B_ * C_ * N_) / 256, 256, 0, stream>>>(G, b_I, b_F, b_O, b_C0, b_C1,
                                                             (float*)d_out);
    (void)in_sizes; (void)n_in; (void)out_size; (void)ws_size;
}
